// SinkhornDistance_another_one_to_multi_12627203850931
// MI455X (gfx1250) — compile-verified, matrix operand order checked by reading, not run on hardware
//
#include <hip/hip_runtime.h>
#include <hip/hip_bf16.h>

typedef _Float16 half8 __attribute__((ext_vector_type(8)));
typedef _Float16 v16h  __attribute__((ext_vector_type(16)));
typedef float    v8f   __attribute__((ext_vector_type(8)));

#define NB 8
#define P1 1024
#define P2 1024
#define DD 256
#define INV_EPS 10.0f
#define EPS_REG 0.1f

// ---------------------------------------------------------------------------
// Zero u, v, errRows and the done flag (all live in d_ws).
// ---------------------------------------------------------------------------
__global__ void init_kernel(float* __restrict__ buf, int n, int* __restrict__ done) {
    int i = blockIdx.x * blockDim.x + threadIdx.x;
    if (i < n) buf[i] = 0.0f;
    if (i == 0) *done = 0;
}

// ---------------------------------------------------------------------------
// Row-normalize x and y (clamped norm), convert to f16. One wave per row.
// ---------------------------------------------------------------------------
__global__ void __launch_bounds__(256)
normalize_kernel(const float* __restrict__ x, const float* __restrict__ y,
                 _Float16* __restrict__ xn, _Float16* __restrict__ yn) {
    const int lane = threadIdx.x & 31;
    const int row  = blockIdx.x * 8 + (threadIdx.x >> 5);   // 0 .. 2*NB*P1-1
    const int NX   = NB * P1;
    const float* src;
    _Float16*    dst;
    if (row < NX) { src = x + (size_t)row * DD;        dst = xn + (size_t)row * DD; }
    else          { src = y + (size_t)(row - NX) * DD; dst = yn + (size_t)(row - NX) * DD; }

    float4 f0 = *(const float4*)(src + lane * 8);
    float4 f1 = *(const float4*)(src + lane * 8 + 4);
    float ss = f0.x*f0.x + f0.y*f0.y + f0.z*f0.z + f0.w*f0.w
             + f1.x*f1.x + f1.y*f1.y + f1.z*f1.z + f1.w*f1.w;
    #pragma unroll
    for (int off = 16; off >= 1; off >>= 1) ss += __shfl_xor(ss, off, 32);
    float scale = 1.0f / fmaxf(sqrtf(ss), 1e-8f);

    half8 o;
    o[0] = (_Float16)(f0.x * scale); o[1] = (_Float16)(f0.y * scale);
    o[2] = (_Float16)(f0.z * scale); o[3] = (_Float16)(f0.w * scale);
    o[4] = (_Float16)(f1.x * scale); o[5] = (_Float16)(f1.y * scale);
    o[6] = (_Float16)(f1.z * scale); o[7] = (_Float16)(f1.w * scale);
    *(half8*)(dst + lane * 8) = o;
}

// ---------------------------------------------------------------------------
// Fragment loader for v_wmma_f32_16x16x32_f16.
// 16-bit A (16x32 MxK) layout: lanes 0-15 -> M=lane, K 0..7 (v0..3) & 16..23
// (v4..7); lanes 16-31 -> M=lane-16, K 8..15 & 24..31. B(32x16 KxN) for
// C = X*Y^T uses the identical per-lane layout on Y's rows.
// ---------------------------------------------------------------------------
__device__ __forceinline__ v16h load_frag(const _Float16* __restrict__ base,
                                          int row, int k0, int lane) {
    const _Float16* p = base + (size_t)(row + (lane & 15)) * DD + k0 + ((lane >> 4) << 3);
    half8 lo = *(const half8*)p;
    half8 hi = *(const half8*)(p + 16);
    return __builtin_shufflevector(lo, hi, 0,1,2,3,4,5,6,7,8,9,10,11,12,13,14,15);
}

// ---------------------------------------------------------------------------
// C = 1 - xn * yn^T per batch. 256 threads = 8 waves; block tile 128x128;
// each wave owns 64x32 = 4x2 WMMA tiles; K loop over 256 in steps of 32.
// 64 v_wmma_f32_16x16x32_f16 per wave.
// ---------------------------------------------------------------------------
__global__ void __launch_bounds__(256)
cost_gemm_kernel(const _Float16* __restrict__ xn, const _Float16* __restrict__ yn,
                 float* __restrict__ C) {
    const int lane = threadIdx.x & 31;
    const int w    = threadIdx.x >> 5;
    const int wm   = w & 1;          // 2 waves in M
    const int wn   = w >> 1;         // 4 waves in N
    const int n    = blockIdx.z;
    const int mBase = blockIdx.y * 128 + wm * 64;
    const int nBase = blockIdx.x * 128 + wn * 32;
    const _Float16* X = xn + (size_t)n * (P1 * DD);
    const _Float16* Y = yn + (size_t)n * (P2 * DD);

    v8f acc[4][2] = {};
    for (int k0 = 0; k0 < DD; k0 += 32) {
        v16h a[4], b[2];
        #pragma unroll
        for (int mt = 0; mt < 4; ++mt) a[mt] = load_frag(X, mBase + mt * 16, k0, lane);
        #pragma unroll
        for (int nt = 0; nt < 2; ++nt) b[nt] = load_frag(Y, nBase + nt * 16, k0, lane);
        #pragma unroll
        for (int mt = 0; mt < 4; ++mt)
            #pragma unroll
            for (int nt = 0; nt < 2; ++nt)
                acc[mt][nt] = __builtin_amdgcn_wmma_f32_16x16x32_f16(
                    false, a[mt], false, b[nt], (short)0, acc[mt][nt], false, false);
    }

    // D layout: lane<16 -> N=lane, rows M=v; lane>=16 -> N=lane-16, rows M=v+8
    float* Cb = C + (size_t)n * (P1 * P2);
    const int colL = lane & 15;
    const int rOff = (lane >> 4) << 3;
    #pragma unroll
    for (int mt = 0; mt < 4; ++mt)
        #pragma unroll
        for (int nt = 0; nt < 2; ++nt) {
            const int colG = nBase + nt * 16 + colL;
            const int rowG = mBase + mt * 16 + rOff;
            #pragma unroll
            for (int r = 0; r < 8; ++r)
                Cb[(size_t)(rowG + r) * P2 + colG] = 1.0f - acc[mt][nt][r];
        }
}

// ---------------------------------------------------------------------------
// u_new[n,i] = eps*(log(mu+1e-8) - lse_j((v_j - C_ij)/eps)). One wave per row,
// streaming online logsumexp, shfl-xor merge. Also emits |u_new - u_old|.
// ---------------------------------------------------------------------------
__global__ void __launch_bounds__(256)
u_step_kernel(const float* __restrict__ C, const float* __restrict__ v,
              const float* __restrict__ mu, float* __restrict__ u,
              float* __restrict__ errRows, const int* __restrict__ done) {
    if (*done) return;
    const int lane = threadIdx.x & 31;
    const int row  = blockIdx.x * 8 + (threadIdx.x >> 5);    // 0..NB*P1-1
    const float* Crow = C + (size_t)row * P2;
    const float* vb   = v + (row >> 10) * P2;

    float m = (vb[lane] - Crow[lane]) * INV_EPS;
    float s = 1.0f;
    for (int j = lane + 32; j < P2; j += 32) {
        float xv = (vb[j] - Crow[j]) * INV_EPS;
        float mn = fmaxf(m, xv);
        s = s * expf(m - mn) + expf(xv - mn);
        m = mn;
    }
    #pragma unroll
    for (int off = 16; off >= 1; off >>= 1) {
        float m2 = __shfl_xor(m, off, 32);
        float s2 = __shfl_xor(s, off, 32);
        float mn = fmaxf(m, m2);
        s = s * expf(m - mn) + s2 * expf(m2 - mn);
        m = mn;
    }
    if (lane == 0) {
        float lse = m + logf(s);
        float un  = EPS_REG * (logf(mu[row] + 1e-8f) - lse);
        errRows[row] = fabsf(un - u[row]);
        u[row] = un;
    }
}

// ---------------------------------------------------------------------------
// v_new[n,j] = eps*(log(1/P2+1e-8) - lse_i((u_i - C_ij)/eps)). 32-column
// tiles, (32,8) block: coalesced 128B row reads, LDS merge across ty.
// ---------------------------------------------------------------------------
__global__ void v_step_kernel(const float* __restrict__ C, const float* __restrict__ u,
                              float* __restrict__ v, const int* __restrict__ done) {
    if (*done) return;
    const int tx = threadIdx.x, ty = threadIdx.y;
    const int n  = blockIdx.y;
    const int j  = blockIdx.x * 32 + tx;
    const float* Cb = C + (size_t)n * (P1 * P2);
    const float* ub = u + n * P1;

    float m = (ub[ty] - Cb[(size_t)ty * P2 + j]) * INV_EPS;
    float s = 1.0f;
    for (int i = ty + 8; i < P1; i += 8) {
        float xv = (ub[i] - Cb[(size_t)i * P2 + j]) * INV_EPS;
        float mn = fmaxf(m, xv);
        s = s * expf(m - mn) + expf(xv - mn);
        m = mn;
    }
    __shared__ float smM[8][32], smS[8][32];
    smM[ty][tx] = m; smS[ty][tx] = s;
    __syncthreads();
    if (ty == 0) {
        #pragma unroll
        for (int k = 1; k < 8; ++k) {
            float m2 = smM[k][tx], s2 = smS[k][tx];
            float mn = fmaxf(m, m2);
            s = s * expf(m - mn) + s2 * expf(m2 - mn);
            m = mn;
        }
        float lse = m + logf(s);
        const float logNu = logf(1.0f / (float)P2 + 1e-8f);
        v[n * P2 + j] = EPS_REG * (logNu - lse);
    }
}

// ---------------------------------------------------------------------------
// err = sum(|du|)/NB ; sticky done flag. Single block, deterministic tree sum.
// ---------------------------------------------------------------------------
__global__ void err_reduce_kernel(const float* __restrict__ errRows, int* __restrict__ done) {
    if (*done) return;
    __shared__ float sm[256];
    float s = 0.0f;
    for (int i = threadIdx.x; i < NB * P1; i += 256) s += errRows[i];
    sm[threadIdx.x] = s;
    __syncthreads();
    for (int st = 128; st >= 1; st >>= 1) {
        if ((int)threadIdx.x < st) sm[threadIdx.x] += sm[threadIdx.x + st];
        __syncthreads();
    }
    if (threadIdx.x == 0 && (sm[0] * (1.0f / NB)) < 0.1f) *done = 1;
}

// ---------------------------------------------------------------------------
// pi = exp((u_i + v_j - C_ij)/eps) stored; cost[n,j] = sum_i pi*C (fused).
// ---------------------------------------------------------------------------
__global__ void pi_cost_kernel(const float* __restrict__ C, const float* __restrict__ u,
                               const float* __restrict__ v, float* __restrict__ pi,
                               float* __restrict__ cost) {
    const int tx = threadIdx.x, ty = threadIdx.y;
    const int n  = blockIdx.y;
    const int j  = blockIdx.x * 32 + tx;
    const float* Cb = C + (size_t)n * (P1 * P2);
    const float* ub = u + n * P1;
    const float  vj = v[n * P2 + j];
    float* pib = pi + (size_t)n * (P1 * P2);

    float acc = 0.0f;
    for (int i = ty; i < P1; i += 8) {
        float c = Cb[(size_t)i * P2 + j];
        float p = expf((ub[i] + vj - c) * INV_EPS);
        pib[(size_t)i * P2 + j] = p;
        acc += p * c;
    }
    __shared__ float sm[8][32];
    sm[ty][tx] = acc;
    __syncthreads();
    if (ty == 0) {
        float t = acc;
        #pragma unroll
        for (int k = 1; k < 8; ++k) t += sm[k][tx];
        cost[n * P2 + j] = t;
    }
}

// ---------------------------------------------------------------------------
extern "C" void kernel_launch(void* const* d_in, const int* in_sizes, int n_in,
                              void* d_out, int out_size, void* d_ws, size_t ws_size,
                              hipStream_t stream) {
    const float* x  = (const float*)d_in[0];   // [8,1024,256]
    const float* y  = (const float*)d_in[1];   // [8,1024,256]
    const float* mu = (const float*)d_in[2];   // [8,1024]

    float* out  = (float*)d_out;
    float* cost = out;                                   // NB*P2
    float* pi   = out + (size_t)NB * P2;                 // NB*P1*P2
    float* Cm   = out + (size_t)NB * P2 + (size_t)NB * P1 * P2;  // NB*P1*P2

    // workspace layout
    char* ws = (char*)d_ws;
    const size_t halfCount = (size_t)NB * P1 * DD;       // 2,097,152 halves each
    _Float16* xn = (_Float16*)ws;
    _Float16* yn = xn + halfCount;
    float* u       = (float*)(ws + 2 * halfCount * sizeof(_Float16));
    float* v       = u + NB * P1;
    float* errRows = v + NB * P2;
    int*   done    = (int*)(errRows + NB * P1);

    // 1) zero u, v, errRows, done
    const int zn = NB * P1 + NB * P2 + NB * P1;          // 24576 floats
    init_kernel<<<(zn + 255) / 256, 256, 0, stream>>>(u, zn, done);

    // 2) normalize + f16 convert
    normalize_kernel<<<(2 * NB * P1) / 8, 256, 0, stream>>>(x, y, xn, yn);

    // 3) cost matrix via WMMA, written straight into the C output region
    cost_gemm_kernel<<<dim3(P2 / 128, P1 / 128, NB), 256, 0, stream>>>(xn, yn, Cm);

    // 4) Sinkhorn iterations (done-gated like the reference scan body)
    for (int it = 0; it < 20; ++it) {
        u_step_kernel<<<(NB * P1) / 8, 256, 0, stream>>>(Cm, v, mu, u, errRows, done);
        v_step_kernel<<<dim3(P2 / 32, NB), dim3(32, 8), 0, stream>>>(Cm, u, v, done);
        err_reduce_kernel<<<1, 256, 0, stream>>>(errRows, done);
    }

    // 5) transport plan + cost
    pi_cost_kernel<<<dim3(P2 / 32, NB), dim3(32, 8), 0, stream>>>(Cm, u, v, pi, cost);
}
